// VQVAE_6700148982572
// MI455X (gfx1250) — compile-verified, matrix operand order checked by reading, not run on hardware
//
#include <hip/hip_runtime.h>
#include <hip/hip_bf16.h>
#include <cstdint>

typedef __bf16 bf16;
typedef bf16 v16bf __attribute__((ext_vector_type(16)));
typedef bf16 v8bf  __attribute__((ext_vector_type(8)));
typedef float v8f  __attribute__((ext_vector_type(8)));

#define FRAME_C 263
#define FRAME_P 288          // ceil32(263)
#define WIDTH_C 512
#define NB_CODE 1024
#define BATCH_C 256
#define T0_C    64
#define XOUT_N  (256L * 64L * 263L)   // 4308992

__device__ __forceinline__ bf16 f2bf(float f) {
  unsigned u = __builtin_bit_cast(unsigned, f);
  unsigned r = (u + 0x7FFFu + ((u >> 16) & 1u)) >> 16;
  unsigned short s = (unsigned short)r;
  return __builtin_bit_cast(bf16, s);
}
__device__ __forceinline__ float bf2f(bf16 h) {
  unsigned short s = __builtin_bit_cast(unsigned short, h);
  unsigned u = ((unsigned)s) << 16;
  return __builtin_bit_cast(float, u);
}
// packed ReLU on two bf16 halves of a dword: clear any half with sign bit set
__device__ __forceinline__ unsigned relu_pk(unsigned u) {
  return u & ~(((u >> 15) & 0x00010001u) * 0xFFFFu);
}
__device__ __forceinline__ uint4 relu_pk4(uint4 q) {
  q.x = relu_pk(q.x); q.y = relu_pk(q.y); q.z = relu_pk(q.z); q.w = relu_pk(q.w);
  return q;
}

// ---------------------------------------------------------------------------
// small utility kernels
// ---------------------------------------------------------------------------
__global__ void zero_k(float* p, int n) {
  int i = blockIdx.x * blockDim.x + threadIdx.x;
  if (i < n) p[i] = 0.f;
}

// tap-major padded weights: o[r, kw*Cinp + ci] = w[(r*Cin + ci)*Kw + kw]
__global__ void convert_w_k(const float* __restrict__ w, bf16* __restrict__ o,
                            int Cout, int Cin, int Kw, int Cinp, long n) {
  const int CKp = Kw * Cinp;
  long i = (long)blockIdx.x * blockDim.x + threadIdx.x;
  long st = (long)gridDim.x * blockDim.x;
  for (; i < n; i += st) {
    int r  = (int)(i / CKp);
    int c  = (int)(i - (long)r * CKp);
    int kw = c / Cinp;
    int ci = c - kw * Cinp;
    float v = (r < Cout && ci < Cin) ? w[((long)r * Cin + ci) * Kw + kw] : 0.f;
    o[i] = f2bf(v);
  }
}

// x (B*T, 263) f32 -> o (B*T, 288) bf16, channel-padded with zeros
__global__ void pad_x_k(const float* __restrict__ x, bf16* __restrict__ o, long n) {
  long i = (long)blockIdx.x * blockDim.x + threadIdx.x;
  long st = (long)gridDim.x * blockDim.x;
  for (; i < n; i += st) {
    int row = (int)(i / FRAME_P);
    int c = (int)(i - (long)row * FRAME_P);
    float v = (c < FRAME_C) ? x[(long)row * FRAME_C + c] : 0.f;
    o[i] = f2bf(v);
  }
}

// ---------------------------------------------------------------------------
// WMMA conv-as-GEMM kernel, channel-last activations.
//   act:  (B, Tin, Cinp) bf16, channel-padded
//   wgt:  (Mp, Kw*Cinp) bf16, tap-major, zero padded rows/channels
//   out:  (B, Tout, Cout) bf16 and/or f32, row-major
// K loop is split: outer over conv taps kw (tap geometry computed once per
// tap), inner over 32-wide channel chunks (pure vector copies). 256 thr =
// 8 waves, 128x128x32 tile, wave grid 4(M) x 2(N), 2x4 WMMA frags per wave.
// ---------------------------------------------------------------------------
#define BM 128
#define BN 128
#define BKK 32
#define LSTR 40   // LDS row stride (elems) -> 80B rows, 16B-aligned frag reads

__global__ __launch_bounds__(256)
void conv_gemm_k(const bf16* __restrict__ act, const bf16* __restrict__ wgt,
                 const float* __restrict__ bias, const bf16* __restrict__ resid,
                 bf16* __restrict__ outb, float* __restrict__ outf,
                 int Cinp, int Tin, int Cout, int Tout,
                 int Kw, int stride, int pad, int dil,
                 int relu_in, int relu_out, int ups)
{
  __shared__ bf16 As[BM * LSTR];
  __shared__ bf16 Bs[BN * LSTR];

  const int tid  = threadIdx.x;
  const int lane = tid & 31;
  const int wave = tid >> 5;
  const int wm   = wave >> 1;   // 0..3
  const int wn   = wave & 1;    // 0..1
  const int m0   = blockIdx.y * BM;
  const int n0   = blockIdx.x * BN;
  const int Ntot = BATCH_C * Tout;
  const int Tup  = Tin * 2;
  const int CKp  = Kw * Cinp;

  v8f acc[2][4] = {};

  // A loader: each thread fills one 16-elem half-row of the weight tile
  const int a_row = tid >> 1;
  const int a_kh  = (tid & 1) * 16;
  // B loader: each thread fills one 16-wide K-chunk for one column n
  const int b_nl = tid & 127;
  const int b_k0 = (tid >> 7) * 16;
  const int n_g  = n0 + b_nl;
  const bool nvalid = n_g < Ntot;
  int bb = 0, to = 0;
  if (nvalid) { bb = n_g / Tout; to = n_g - bb * Tout; }

  for (int kw = 0; kw < Kw; ++kw) {
    // ---- per-tap geometry (uniform across the whole inner K loop) ----
    int tin = 0; bool tv = nvalid;
    if (ups) { int tu = to + kw * dil - pad; tv = tv && (tu >= 0) && (tu < Tup); tin = tu >> 1; }
    else     { tin = to * stride + kw * dil - pad; tv = tv && (tin >= 0) && (tin < Tin); }
    if (!tv) tin = 0;
    const bf16* asrc = wgt + (size_t)(m0 + a_row) * CKp + kw * Cinp + a_kh;
    const bf16* bsrc = act + ((size_t)bb * Tin + tin) * Cinp + b_k0;

    for (int ci = 0; ci < Cinp; ci += BKK) {
      // A tile: pure vector copies (fully padded)
      *(uint4*)(As + a_row * LSTR + a_kh)     = *(const uint4*)(asrc + ci);
      *(uint4*)(As + a_row * LSTR + a_kh + 8) = *(const uint4*)(asrc + ci + 8);
      // B tile: contiguous channel chunk, loop-invariant predicate
      uint4 lo = {0u, 0u, 0u, 0u}, hi = {0u, 0u, 0u, 0u};
      if (tv) {
        lo = *(const uint4*)(bsrc + ci);
        hi = *(const uint4*)(bsrc + ci + 8);
        if (relu_in) { lo = relu_pk4(lo); hi = relu_pk4(hi); }
      }
      *(uint4*)(Bs + b_nl * LSTR + b_k0)     = lo;
      *(uint4*)(Bs + b_nl * LSTR + b_k0 + 8) = hi;
      __syncthreads();

      // ---- compute: 2x4 WMMA fragments per wave ----
      const int kb   = ((lane >> 4) & 1) * 8;
      const int mrow = lane & 15;
      union AB { v16bf v; v8bf h[2]; };
      AB a[2], b[4];
      #pragma unroll
      for (int i = 0; i < 2; ++i) {
        const bf16* p = As + (wm * 32 + i * 16 + mrow) * LSTR + kb;
        a[i].h[0] = *(const v8bf*)p;
        a[i].h[1] = *(const v8bf*)(p + 16);
      }
      #pragma unroll
      for (int j = 0; j < 4; ++j) {
        const bf16* p = Bs + (wn * 64 + j * 16 + mrow) * LSTR + kb;
        b[j].h[0] = *(const v8bf*)p;
        b[j].h[1] = *(const v8bf*)(p + 16);
      }
      #pragma unroll
      for (int i = 0; i < 2; ++i)
        #pragma unroll
        for (int j = 0; j < 4; ++j)
          acc[i][j] = __builtin_amdgcn_wmma_f32_16x16x32_bf16(
              false, a[i].v, false, b[j].v, (short)0, acc[i][j], false, false);
      __syncthreads();
    }
  }

  // ---- epilogue: bias / relu / residual / stores (channel-last) ----
  #pragma unroll
  for (int i = 0; i < 2; ++i) {
    #pragma unroll
    for (int j = 0; j < 4; ++j) {
      #pragma unroll
      for (int r = 0; r < 8; ++r) {
        int mg = m0 + wm * 32 + i * 16 + r + ((lane >> 4) << 3);
        int ng = n0 + wn * 64 + j * 16 + (lane & 15);
        if (mg < Cout && ng < Ntot) {
          float v = acc[i][j][r];
          if (bias) v += bias[mg];
          if (relu_out && v < 0.f) v = 0.f;
          if (resid) v += bf2f(resid[(size_t)ng * Cout + mg]);
          if (outb) outb[(size_t)ng * Cout + mg] = f2bf(v);
          if (outf) outf[(size_t)ng * Cout + mg] = v;
        }
      }
    }
  }
}

// ---------------------------------------------------------------------------
// VQ kernels (row-major layouts: row = b*8+t)
// ---------------------------------------------------------------------------
__global__ void c2_k(const float* __restrict__ cb, float* __restrict__ c2) {
  int c = blockIdx.x * blockDim.x + threadIdx.x;
  if (c >= NB_CODE) return;
  const float* p = cb + (long)c * WIDTH_C;
  float s = 0.f;
  for (int k = 0; k < WIDTH_C; ++k) s += p[k] * p[k];
  c2[c] = s;
}

// dots (row, NB_CODE). dist-equivalent = c2[c] - 2*dot (z-norm constant per row)
__global__ void argmin_k(const float* __restrict__ dots, const float* __restrict__ c2,
                         int* __restrict__ idx, float* __restrict__ counts) {
  int n = blockIdx.x * blockDim.x + threadIdx.x;
  if (n >= BATCH_C * 8) return;
  const float* dp = dots + (size_t)n * NB_CODE;
  float best = 3.4e38f; int bi = 0;
  for (int c = 0; c < NB_CODE; ++c) {
    float d = c2[c] - 2.f * dp[c];
    if (d < best) { best = d; bi = c; }
  }
  idx[n] = bi;
  atomicAdd(counts + bi, 1.0f);
}

// z_q forward value = codebook[idx]; layout (row, C) bf16
__global__ void build_zq_k(const float* __restrict__ cb, const int* __restrict__ idx,
                           bf16* __restrict__ zq, long n) {
  long e = (long)blockIdx.x * blockDim.x + threadIdx.x;
  if (e >= n) return;
  int row = (int)(e >> 9);         // / 512
  int c   = (int)(e & 511);
  zq[e] = f2bf(cb[(size_t)idx[row] * WIDTH_C + c]);
}

__global__ void commit_k(const float* __restrict__ z, const float* __restrict__ cb,
                         const int* __restrict__ idx, float* __restrict__ out) {
  __shared__ float red[256];
  const long total = (long)BATCH_C * 8 * WIDTH_C;   // 1048576
  long i = (long)blockIdx.x * blockDim.x + threadIdx.x;
  long st = (long)gridDim.x * blockDim.x;
  float s = 0.f;
  for (long e = i; e < total; e += st) {
    int row = (int)(e >> 9);
    int c   = (int)(e & 511);
    float d = z[e] - cb[(size_t)idx[row] * WIDTH_C + c];
    s += d * d;
  }
  red[threadIdx.x] = s;
  __syncthreads();
  for (int k = 128; k > 0; k >>= 1) {
    if (threadIdx.x < k) red[threadIdx.x] += red[threadIdx.x + k];
    __syncthreads();
  }
  if (threadIdx.x == 0) atomicAdd(out, red[0] / (float)total);
}

__global__ void perp_k(const float* __restrict__ counts, float* __restrict__ out) {
  __shared__ float red[1024];
  int i = threadIdx.x;
  float p = counts[i] / (float)(BATCH_C * 8);
  red[i] = p * logf(p + 1e-7f);
  __syncthreads();
  for (int k = 512; k > 0; k >>= 1) {
    if (i < k) red[i] += red[i + k];
    __syncthreads();
  }
  if (i == 0) out[0] = expf(-red[0]);
}

// ---------------------------------------------------------------------------
// host orchestration
// ---------------------------------------------------------------------------
extern "C" void kernel_launch(void* const* d_in, const int* in_sizes, int n_in,
                              void* d_out, int out_size, void* d_ws, size_t ws_size,
                              hipStream_t stream) {
  (void)in_sizes; (void)n_in; (void)out_size; (void)ws_size;

  // jax tree_leaves order (dict keys sorted): codebook, dec_params, enc_params, x
  // dec: b_in(1) b_mid(2) b_out(3) stages[3]{bu, res[3]{b1,b2,w1,w2}, wu} w_in(46) w_mid(47) w_out(48)
  // enc: b_in(49) b_out(50) stages[3]{bd, res[3]{b1,b2,w1,w2}, wd} w_in(93) w_out(94)  x(95)
  auto fin = [&](int i) -> const float* { return (const float*)d_in[i]; };
  const float* codebook = fin(0);
  const float* x_in = fin(95);
  float* out_f = (float*)d_out;

  char* ws = (char*)d_ws;
  size_t off = 0;
  auto alloc = [&](size_t bytes) -> void* {
    void* p = ws + off;
    off = (off + bytes + 255) & ~(size_t)255;
    return p;
  };

  auto prep_w = [&](int leaf, int Cout, int Cin, int Kw) -> bf16* {
    int Cinp = (Cin + 31) & ~31;
    int CKp = Kw * Cinp;
    int Mp = (Cout + 127) & ~127;
    long n = (long)Mp * CKp;
    bf16* o = (bf16*)alloc((size_t)n * sizeof(bf16));
    int blocks = (int)((n + 255) / 256);
    if (blocks > 16384) blocks = 16384;
    convert_w_k<<<blocks, 256, 0, stream>>>(fin(leaf), o, Cout, Cin, Kw, Cinp, n);
    return o;
  };

  // --- weight prep (bf16, tap-major, padded) ---
  bf16* w_enc_in = prep_w(93, WIDTH_C, FRAME_C, 3);
  bf16* w_enc_out = prep_w(94, WIDTH_C, WIDTH_C, 3);
  bf16* w_enc_wd[3]; bf16* w_enc_w1[3][3]; bf16* w_enc_w2[3][3];
  for (int s = 0; s < 3; ++s) {
    int base = 51 + s * 14;
    w_enc_wd[s] = prep_w(base + 13, WIDTH_C, WIDTH_C, 4);
    for (int j = 0; j < 3; ++j) {
      w_enc_w1[s][j] = prep_w(base + 3 + 4 * j, WIDTH_C, WIDTH_C, 3);
      w_enc_w2[s][j] = prep_w(base + 4 + 4 * j, WIDTH_C, WIDTH_C, 1);
    }
  }
  bf16* w_dec_in = prep_w(46, WIDTH_C, WIDTH_C, 3);
  bf16* w_dec_mid = prep_w(47, WIDTH_C, WIDTH_C, 3);
  bf16* w_dec_out = prep_w(48, FRAME_C, WIDTH_C, 3);
  bf16* w_dec_wu[3]; bf16* w_dec_w1[3][3]; bf16* w_dec_w2[3][3];
  for (int s = 0; s < 3; ++s) {
    int base = 4 + s * 14;
    w_dec_wu[s] = prep_w(base + 13, WIDTH_C, WIDTH_C, 3);
    for (int j = 0; j < 3; ++j) {
      w_dec_w1[s][j] = prep_w(base + 3 + 4 * j, WIDTH_C, WIDTH_C, 3);
      w_dec_w2[s][j] = prep_w(base + 4 + 4 * j, WIDTH_C, WIDTH_C, 1);
    }
  }
  bf16* cb_bf = prep_w(0, NB_CODE, WIDTH_C, 1);

  // --- activation / scratch buffers (channel-last) ---
  const size_t ACT_MAX = (size_t)BATCH_C * T0_C * WIDTH_C;    // 8.4M elems
  bf16* bufs[3];
  for (int i = 0; i < 3; ++i) bufs[i] = (bf16*)alloc(ACT_MAX * sizeof(bf16));
  float* zf32 = (float*)alloc((size_t)BATCH_C * 8 * WIDTH_C * sizeof(float));
  bf16* zbf = (bf16*)alloc((size_t)BATCH_C * 8 * WIDTH_C * sizeof(bf16));
  float* dots = (float*)alloc((size_t)BATCH_C * 8 * NB_CODE * sizeof(float));
  int* idxb = (int*)alloc((size_t)BATCH_C * 8 * sizeof(int));
  float* counts = (float*)alloc(NB_CODE * sizeof(float));
  float* c2b = (float*)alloc(NB_CODE * sizeof(float));

  zero_k<<<4, 256, 0, stream>>>(counts, NB_CODE);
  zero_k<<<1, 2, 0, stream>>>(out_f + XOUT_N, 2);

  auto conv = [&](const bf16* act, const bf16* w, const float* bias, const bf16* resid,
                  bf16* outb, float* outf, int Cinp, int Tin, int Cout, int Tout,
                  int Kw, int stride, int pad, int dil,
                  int ri, int ro, int ups) {
    int Ntot = BATCH_C * Tout;
    dim3 g((Ntot + BN - 1) / BN, (Cout + BM - 1) / BM);
    conv_gemm_k<<<g, 256, 0, stream>>>(act, w, bias, resid, outb, outf,
                                       Cinp, Tin, Cout, Tout,
                                       Kw, stride, pad, dil, ri, ro, ups);
  };

  const int dils[3] = {1, 3, 9};

  // =================== ENCODER ===================
  pad_x_k<<<8192, 256, 0, stream>>>(x_in, bufs[0], (long)BATCH_C * T0_C * FRAME_P);
  // h = relu(conv_in(x))
  conv(bufs[0], w_enc_in, fin(49), nullptr, bufs[1], nullptr,
       FRAME_P, T0_C, WIDTH_C, T0_C, 3, 1, 1, 1, /*ri*/0, /*ro*/1, 0);
  int ia = 1, ib = 0, ic = 2, Tcur = T0_C;
  for (int s = 0; s < 3; ++s) {
    int base = 51 + s * 14;
    // downsample K=4,s=2,p=1
    conv(bufs[ia], w_enc_wd[s], fin(base), nullptr, bufs[ib], nullptr,
         WIDTH_C, Tcur, WIDTH_C, Tcur / 2, 4, 2, 1, 1, 0, 0, 0);
    Tcur /= 2; { int t = ia; ia = ib; ib = t; }
    for (int j = 0; j < 3; ++j) {
      int d = dils[j];
      conv(bufs[ia], w_enc_w1[s][j], fin(base + 1 + 4 * j), nullptr, bufs[ib], nullptr,
           WIDTH_C, Tcur, WIDTH_C, Tcur, 3, 1, d, d, /*ri*/1, 0, 0);
      conv(bufs[ib], w_enc_w2[s][j], fin(base + 2 + 4 * j), bufs[ia], bufs[ic], nullptr,
           WIDTH_C, Tcur, WIDTH_C, Tcur, 1, 1, 0, 1, /*ri*/1, 0, 0);
      { int t = ia; ia = ic; ic = t; }
    }
  }
  // z = conv_out(h): write f32 + bf16 (both row-major (n, C))
  conv(bufs[ia], w_enc_out, fin(50), nullptr, zbf, zf32,
       WIDTH_C, 8, WIDTH_C, 8, 3, 1, 1, 1, 0, 0, 0);

  // =================== VQ ===================
  c2_k<<<4, 256, 0, stream>>>(codebook, c2b);
  // dots = z . codebook^T via WMMA GEMM (Kw=1 "conv"), row-major (n, code)
  conv(zbf, cb_bf, nullptr, nullptr, nullptr, dots,
       WIDTH_C, 8, NB_CODE, 8, 1, 1, 0, 1, 0, 0, 0);
  argmin_k<<<8, 256, 0, stream>>>(dots, c2b, idxb, counts);
  build_zq_k<<<4096, 256, 0, stream>>>(codebook, idxb, bufs[ib], (long)BATCH_C * 8 * WIDTH_C);
  commit_k<<<512, 256, 0, stream>>>(zf32, codebook, idxb, out_f + XOUT_N);
  perp_k<<<1, 1024, 0, stream>>>(counts, out_f + XOUT_N + 1);

  // =================== DECODER ===================
  int ha = ia, hb = ib, hc = ic;   // hb holds z_q
  conv(bufs[hb], w_dec_in, fin(1), nullptr, bufs[ha], nullptr,
       WIDTH_C, 8, WIDTH_C, 8, 3, 1, 1, 1, 0, /*ro*/1, 0);
  Tcur = 8;
  for (int s = 0; s < 3; ++s) {
    int base = 4 + s * 14;
    for (int j = 0; j < 3; ++j) {
      int d = dils[2 - j];   // 9,3,1
      conv(bufs[ha], w_dec_w1[s][j], fin(base + 1 + 4 * j), nullptr, bufs[hb], nullptr,
           WIDTH_C, Tcur, WIDTH_C, Tcur, 3, 1, d, d, /*ri*/1, 0, 0);
      conv(bufs[hb], w_dec_w2[s][j], fin(base + 2 + 4 * j), bufs[ha], bufs[hc], nullptr,
           WIDTH_C, Tcur, WIDTH_C, Tcur, 1, 1, 0, 1, /*ri*/1, 0, 0);
      { int t = ha; ha = hc; hc = t; }
    }
    // nearest-x2 upsample fused into conv (K=3, p=1 on upsampled grid)
    conv(bufs[ha], w_dec_wu[s], fin(base), nullptr, bufs[hb], nullptr,
         WIDTH_C, Tcur, WIDTH_C, Tcur * 2, 3, 1, 1, 1, 0, 0, /*ups*/1);
    Tcur *= 2; { int t = ha; ha = hb; hb = t; }
  }
  // h = relu(conv_mid(h))
  conv(bufs[ha], w_dec_mid, fin(2), nullptr, bufs[hb], nullptr,
       WIDTH_C, T0_C, WIDTH_C, T0_C, 3, 1, 1, 1, 0, /*ro*/1, 0);
  // x_out = conv_out(h): f32 row-major (b*T+t, 263) straight into d_out
  conv(bufs[hb], w_dec_out, fin(3), nullptr, nullptr, out_f,
       WIDTH_C, T0_C, FRAME_C, T0_C, 3, 1, 1, 1, 0, 0, 0);
}
